// GraphTemporal_89756226552319
// MI455X (gfx1250) — matrix operand order repaired and checked by reading, not compiled, not run on hardware
//
#include <hip/hip_runtime.h>
#include <hip/hip_bf16.h>
#include <math.h>

typedef __bf16 bf16_t;
typedef __attribute__((ext_vector_type(16))) __bf16   v16bf;
typedef __attribute__((ext_vector_type(8)))  float    v8f;
typedef __attribute__((ext_vector_type(4)))  unsigned v4u;

#define NB 64
#define HD 64
#define NLAYER 4
#define HROW 136   // padded LDS row stride (elems): 272B = 68 dwords ≡ 4 (mod 64 banks)

// ---------- helpers ----------

__device__ __forceinline__ bf16_t f2bf(float f) {
  return (bf16_t)f;   // native f32->bf16 (RNE) — lowers to v_cvt*bf16_f32 on gfx1250
}

// order-preserving float<->uint encoding for signed atomic max
__device__ __forceinline__ unsigned fenc(float f) {
  unsigned u = __float_as_uint(f);
  return (u & 0x80000000u) ? ~u : (u ^ 0x80000000u);
}
__device__ __forceinline__ float fdec(unsigned k) {
  unsigned u = (k & 0x80000000u) ? (k ^ 0x80000000u) : ~k;
  return __uint_as_float(u);
}

// Build a v16bf A/B-operand from two contiguous 8-element bf16 runs (16B each)
__device__ __forceinline__ v16bf load_pair16(const bf16_t* p0, const bf16_t* p1) {
  union { v16bf v; v4u u[2]; } t;
  t.u[0] = *(const v4u*)p0;
  t.u[1] = *(const v4u*)p1;
  return t.v;
}

// Same, but source is f32 (converted to bf16 on the fly)
__device__ __forceinline__ v16bf cvt_pair8f(const float* p0, const float* p1) {
  union { v16bf v; bf16_t e[16]; } t;
#pragma unroll
  for (int i = 0; i < 8; ++i) { t.e[i] = f2bf(p0[i]); t.e[8 + i] = f2bf(p1[i]); }
  return t.v;
}

__device__ __forceinline__ v8f wmma_bf16(v16bf a, v16bf b, v8f c) {
  // D = A(16x32) * B(32x16) + C, f32 accumulate
  return __builtin_amdgcn_wmma_f32_16x16x32_bf16(false, a, false, b, (short)0, c,
                                                 false, false);
}

// ---------- weight packing: row-major f32 W[Nout][Kdim] -> WMMA-B bf16 layout ----------
// dst[((kc*NT + nt)*32 + lane)*16 + i] = W[nt*16 + (lane&15)][kc*32 + (lane>>4)*8 + (i&7) + (i>>3)*16]
__global__ void pack_b_kernel(const float* __restrict__ W, bf16_t* __restrict__ dst,
                              int Kdim, int NT, int total) {
  for (int idx = blockIdx.x * blockDim.x + threadIdx.x; idx < total;
       idx += gridDim.x * blockDim.x) {
    int i    = idx & 15;
    int lane = (idx >> 4) & 31;
    int ntkc = idx >> 9;
    int nt   = ntkc % NT;
    int kc   = ntkc / NT;
    dst[idx] = f2bf(W[(nt * 16 + (lane & 15)) * Kdim + (kc * 32 + (lane >> 4) * 8 + (i & 7) + ((i >> 3) << 4))]);
  }
}

// ---------- init kernels ----------

__global__ void fill_u32_kernel(unsigned* __restrict__ p, unsigned val, int total) {
  for (int i = blockIdx.x * blockDim.x + threadIdx.x; i < total;
       i += gridDim.x * blockDim.x)
    p[i] = val;
}

__global__ void node_init_kernel(const float* __restrict__ v0, const float* __restrict__ v1,
                                 const float* __restrict__ wte,
                                 const float* __restrict__ Win0, const float* __restrict__ bin0,
                                 const float* __restrict__ Win1, const float* __restrict__ bin1,
                                 float* __restrict__ xf, bf16_t* __restrict__ xb, int N) {
  int n = blockIdx.x * blockDim.x + threadIdx.x;
  if (n >= N) return;
  float a = v0[n], b = v1[n];
#pragma unroll
  for (int h = 0; h < HD; ++h) {
    float v = wte[h] + wte[HD + h] + bin0[h] + bin1[h] + a * Win0[h] + b * Win1[h];
    xf[(long)n * HD + h] = v;
    xb[(long)n * HD + h] = f2bf(v);
  }
}

__global__ void edge_init_kernel(const float* __restrict__ dist,
                                 const float* __restrict__ Wedge,
                                 const float* __restrict__ bedge,
                                 bf16_t* __restrict__ eb, int E) {
  int e = blockIdx.x * blockDim.x + threadIdx.x;
  if (e >= E) return;
  float d = dist[e];
#pragma unroll
  for (int h = 0; h < HD; ++h)
    eb[(long)e * HD + h] = f2bf(d * Wedge[h] + bedge[h]);
}

// ---------- edge MLP + scatter-max ----------
// msg = relu([x[row] | e] @ W_el^T + b_el); agg[col] = max(agg[col], msg)
// one wave per 16-edge tile; K=128 (4 chunks), Nout=64 (4 tiles); 16 WMMAs/wave
__global__ void __launch_bounds__(256)
edge_mlp_kernel(const bf16_t* __restrict__ xb, const bf16_t* __restrict__ eb,
                const int* __restrict__ rowi, const int* __restrict__ coli,
                const bf16_t* __restrict__ wpk, const float* __restrict__ bias,
                unsigned* __restrict__ agg, int E) {
  int wave = (int)((blockIdx.x * blockDim.x + threadIdx.x) >> 5);
  int lane = threadIdx.x & 31;
  int m = lane & 15, hi = lane >> 4;
  long ebase = (long)wave * 16;
  if (ebase >= E) return;                     // wave-uniform; no barriers in kernel
  int eidx = (int)ebase + m; if (eidx >= E) eidx = E - 1;
  int src = rowi[eidx];
  const bf16_t* xr = xb + (long)src * HD;
  const bf16_t* er = eb + (long)eidx * HD;
  v16bf a[4];
  a[0] = load_pair16(xr + hi * 8,      xr + 16 + hi * 8);
  a[1] = load_pair16(xr + 32 + hi * 8, xr + 48 + hi * 8);
  a[2] = load_pair16(er + hi * 8,      er + 16 + hi * 8);
  a[3] = load_pair16(er + 32 + hi * 8, er + 48 + hi * 8);

  v8f acc[4];
#pragma unroll
  for (int nt = 0; nt < 4; ++nt) {
    v8f c = {};
#pragma unroll
    for (int kc = 0; kc < 4; ++kc) {
      v16bf b = *(const v16bf*)(wpk + (((kc * 4 + nt) * 32 + lane) << 4));
      c = wmma_bf16(a[kc], b, c);
    }
    acc[nt] = c;
  }

  // scatter: C layout row = r + hi*8 (edge), col = lane&15 (feature)
  if (ebase + 16 <= (long)E) {                // uniform fast path: no per-row guards
    int dcol[8];
#pragma unroll
    for (int r = 0; r < 8; ++r) dcol[r] = coli[ebase + r + hi * 8];
#pragma unroll
    for (int nt = 0; nt < 4; ++nt) {
      int n = nt * 16 + m;
      float bs = bias[n];
#pragma unroll
      for (int r = 0; r < 8; ++r) {
        float v = acc[nt][r] + bs;
        v = v > 0.f ? v : 0.f;                // relu >= 0 -> raw-bits u32 max is exact
        atomicMax(agg + (long)dcol[r] * HD + n, __float_as_uint(v));
      }
    }
  } else {                                    // tail tile
    int dcol[8];
#pragma unroll
    for (int r = 0; r < 8; ++r) {
      long ei = ebase + r + hi * 8;
      dcol[r] = (ei < E) ? coli[ei] : -1;
    }
#pragma unroll
    for (int nt = 0; nt < 4; ++nt) {
      int n = nt * 16 + m;
      float bs = bias[n];
#pragma unroll
      for (int r = 0; r < 8; ++r) {
        if (dcol[r] >= 0) {
          float v = acc[nt][r] + bs;
          v = v > 0.f ? v : 0.f;
          atomicMax(agg + (long)dcol[r] * HD + n, __float_as_uint(v));
        }
      }
    }
  }
}

// ---------- fused node MLP: two-phase GEMM1 -> LDS re-layout -> GEMM2 + residual ----------
// h1 = relu([agg | x | u[batch]] @ W_n1^T + b_n1)   (K=192, Nout=256, split in 2x128 cols)
// x  = x + h1 @ W_n2^T + b_n2                        (K=256 accumulated across phases)
// also scatter-max new x into per-graph xu (signed encoding)
__global__ void __launch_bounds__(256)
node_mlp_kernel(const unsigned* __restrict__ agg, const bf16_t* __restrict__ ub,
                const int* __restrict__ batch,
                const bf16_t* __restrict__ w1pk, const float* __restrict__ b1,
                const bf16_t* __restrict__ w2pk, const float* __restrict__ b2,
                float* __restrict__ xf, bf16_t* __restrict__ xb,
                unsigned* __restrict__ xu, int N) {
  __shared__ bf16_t h1s[8 * 16 * HROW];       // ~34 KB: per-wave 16 x 128(+pad) tile
  int w = threadIdx.x >> 5;
  int lane = threadIdx.x & 31;
  int m = lane & 15, hi = lane >> 4;
  long nbase = ((long)blockIdx.x * 8 + w) * 16;
  int nd = (int)nbase + m; if (nd >= N) nd = N - 1;   // clamp; stores are guarded
  const float*  ar = (const float*)agg + (long)nd * HD;  // bits are non-negative floats
  const bf16_t* xr = xb + (long)nd * HD;
  const bf16_t* ur = ub + (long)batch[nd] * HD;
  v16bf a[6];
  a[0] = cvt_pair8f(ar + hi * 8,       ar + 16 + hi * 8);
  a[1] = cvt_pair8f(ar + 32 + hi * 8,  ar + 48 + hi * 8);
  a[2] = load_pair16(xr + hi * 8,      xr + 16 + hi * 8);
  a[3] = load_pair16(xr + 32 + hi * 8, xr + 48 + hi * 8);
  a[4] = load_pair16(ur + hi * 8,      ur + 16 + hi * 8);
  a[5] = load_pair16(ur + 32 + hi * 8, ur + 48 + hi * 8);

  bf16_t* hbase = h1s + w * 16 * HROW;
  v8f c2[4] = {};                             // GEMM2 accumulators, live across phases

#pragma unroll
  for (int half = 0; half < 2; ++half) {
    // GEMM1 for output columns [half*128, half*128+128)
#pragma unroll
    for (int nt8 = 0; nt8 < 8; ++nt8) {
      int nt = half * 8 + nt8;
      v8f c = {};
#pragma unroll
      for (int kc = 0; kc < 6; ++kc) {
        v16bf b = *(const v16bf*)(w1pk + (((kc * 16 + nt) * 32 + lane) << 4));
        c = wmma_bf16(a[kc], b, c);
      }
      int n = nt * 16 + m;
      float bs = b1[n];
      int nl = nt8 * 16 + m;                  // local column within this half
#pragma unroll
      for (int r = 0; r < 8; ++r) {
        float v = c[r] + bs; v = v > 0.f ? v : 0.f;
        hbase[(r + hi * 8) * HROW + nl] = f2bf(v);
      }
    }
    __syncthreads();                          // all waves participate (no early return)

    // GEMM2 partial accumulate over this half's 128 K-values (4 chunks)
    const bf16_t* hr = hbase + m * HROW;      // A layout: row = lane&15
    v16bf ha[4];
#pragma unroll
    for (int kc = 0; kc < 4; ++kc)
      ha[kc] = load_pair16(hr + kc * 32 + hi * 8, hr + kc * 32 + 16 + hi * 8);
#pragma unroll
    for (int nt = 0; nt < 4; ++nt) {
#pragma unroll
      for (int kc = 0; kc < 4; ++kc) {
        int kg = half * 4 + kc;               // global K-chunk index for weights
        v16bf b = *(const v16bf*)(w2pk + (((kg * 4 + nt) * 32 + lane) << 4));
        c2[nt] = wmma_bf16(ha[kc], b, c2[nt]);
      }
    }
    __syncthreads();                          // before next phase overwrites LDS
  }

  // store: residual add + per-graph scatter-max
  if (nbase + 16 <= (long)N) {                // uniform fast path
    int bt[8];
#pragma unroll
    for (int r = 0; r < 8; ++r) bt[r] = batch[nbase + r + hi * 8];
#pragma unroll
    for (int nt = 0; nt < 4; ++nt) {
      int n = nt * 16 + m;
      float bs = b2[n];
#pragma unroll
      for (int r = 0; r < 8; ++r) {
        long idx = (nbase + r + hi * 8) * HD + n;
        float xn = xf[idx] + c2[nt][r] + bs;
        xf[idx] = xn;
        xb[idx] = f2bf(xn);
        atomicMax(xu + (long)bt[r] * HD + n, fenc(xn));
      }
    }
  } else {                                    // tail tile
#pragma unroll
    for (int nt = 0; nt < 4; ++nt) {
      int n = nt * 16 + m;
      float bs = b2[n];
#pragma unroll
      for (int r = 0; r < 8; ++r) {
        long node = nbase + r + hi * 8;
        if (node < N) {
          long idx = node * HD + n;
          float xn = xf[idx] + c2[nt][r] + bs;
          xf[idx] = xn;
          xb[idx] = f2bf(xn);
          atomicMax(xu + (long)batch[node] * HD + n, fenc(xn));
        }
      }
    }
  }
}

// ---------- per-graph global update + LSTM cell (tiny: B=64) ----------
__global__ void global_update_kernel(const unsigned* __restrict__ xu,
                                     float* __restrict__ uf, bf16_t* __restrict__ ub,
                                     const float* __restrict__ Wgl, const float* __restrict__ bgl,
                                     const float* __restrict__ Wih, const float* __restrict__ bih,
                                     const float* __restrict__ bhh,
                                     float* __restrict__ out_cs) {
  __shared__ float xus[HD], us[HD], uns[HD];
  int b = blockIdx.x, t = threadIdx.x;
  float f = fdec(xu[b * HD + t]);
  if ((__float_as_uint(f) & 0x7F800000u) == 0x7F800000u) f = 0.f;  // empty -> 0
  xus[t] = f;
  us[t]  = uf[b * HD + t];
  __syncthreads();
  float s = bgl[t];
  for (int k = 0; k < HD; ++k) s += xus[k] * Wgl[t * 2 * HD + k];
  for (int k = 0; k < HD; ++k) s += us[k]  * Wgl[t * 2 * HD + HD + k];
  float un = us[t] + (s > 0.f ? s : 0.f);
  uns[t] = un;
  __syncthreads();
  float ig = bih[t] + bhh[t];
  float gg = bih[2 * HD + t] + bhh[2 * HD + t];
  float og = bih[3 * HD + t] + bhh[3 * HD + t];
  for (int k = 0; k < HD; ++k) {
    float uk = uns[k];
    ig += uk * Wih[t * HD + k];
    gg += uk * Wih[(2 * HD + t) * HD + k];
    og += uk * Wih[(3 * HD + t) * HD + k];
  }
  float c  = (1.f / (1.f + __expf(-ig))) * tanhf(gg);
  float u2 = (1.f / (1.f + __expf(-og))) * tanhf(c);
  uf[b * HD + t] = u2;
  ub[b * HD + t] = f2bf(u2);
  out_cs[b * HD + t] = c;
}

// ---------- output projections ----------
__global__ void out_proj_kernel(const float* __restrict__ xf,
                                const float* __restrict__ W0, const float* __restrict__ b0,
                                const float* __restrict__ W1, const float* __restrict__ b1,
                                float* __restrict__ out, int N) {
  int n = blockIdx.x * blockDim.x + threadIdx.x;
  if (n >= N) return;
  const float* xr = xf + (long)n * HD;
  float s0 = 0.f, s1 = 0.f;
#pragma unroll
  for (int h = 0; h < HD; ++h) { float v = xr[h]; s0 += v * W0[h]; s1 += v * W1[h]; }
  out[n]     = s0 + b0[0];
  out[N + n] = s1 + b1[0];
}

// ---------- host launcher ----------
extern "C" void kernel_launch(void* const* d_in, const int* in_sizes, int n_in,
                              void* d_out, int out_size, void* d_ws, size_t ws_size,
                              hipStream_t stream) {
  const float* v0    = (const float*)d_in[0];
  const float* v1    = (const float*)d_in[1];
  const float* dist  = (const float*)d_in[2];
  const int*   eidx  = (const int*)d_in[3];
  const int*   batch = (const int*)d_in[4];
  const float* wte   = (const float*)d_in[6];
  const float* Win0  = (const float*)d_in[7];
  const float* bin0  = (const float*)d_in[8];
  const float* Win1  = (const float*)d_in[9];
  const float* bin1  = (const float*)d_in[10];
  const float* Wout0 = (const float*)d_in[11];
  const float* bout0 = (const float*)d_in[12];
  const float* Wout1 = (const float*)d_in[13];
  const float* bout1 = (const float*)d_in[14];
  const float* Wedge = (const float*)d_in[15];
  const float* bedge = (const float*)d_in[16];
  const float* Wel   = (const float*)d_in[17];
  const float* bel   = (const float*)d_in[18];
  const float* Wn1   = (const float*)d_in[19];
  const float* bn1   = (const float*)d_in[20];
  const float* Wn2   = (const float*)d_in[21];
  const float* bn2   = (const float*)d_in[22];
  const float* Wgl   = (const float*)d_in[23];
  const float* bgl   = (const float*)d_in[24];
  const float* Wih   = (const float*)d_in[25];
  const float* bih   = (const float*)d_in[27];
  const float* bhh   = (const float*)d_in[28];
  const int N = in_sizes[0];
  const int E = in_sizes[2];

  // workspace carve (256B aligned)
  char* wsp = (char*)d_ws;
  auto carve = [&](size_t bytes) -> void* {
    void* p = (void*)wsp;
    wsp += (bytes + 255) & ~(size_t)255;
    return p;
  };
  float*    xf    = (float*)carve((size_t)N * HD * 4);
  bf16_t*   xb    = (bf16_t*)carve((size_t)N * HD * 2);
  bf16_t*   eb    = (bf16_t*)carve((size_t)E * HD * 2);
  unsigned* agg   = (unsigned*)carve((size_t)N * HD * 4);
  float*    uf    = (float*)carve((size_t)NB * HD * 4);
  bf16_t*   ub    = (bf16_t*)carve((size_t)NB * HD * 2);
  unsigned* xu    = (unsigned*)carve((size_t)NB * HD * 4);
  bf16_t*   welpk = (bf16_t*)carve((size_t)4 * 4 * 512 * 2);
  bf16_t*   wn1pk = (bf16_t*)carve((size_t)6 * 16 * 512 * 2);
  bf16_t*   wn2pk = (bf16_t*)carve((size_t)8 * 4 * 512 * 2);

  pack_b_kernel<<<32, 256, 0, stream>>>(Wel, welpk, 2 * HD, 4, 4 * 4 * 512);
  pack_b_kernel<<<192, 256, 0, stream>>>(Wn1, wn1pk, 3 * HD, 16, 6 * 16 * 512);
  pack_b_kernel<<<64, 256, 0, stream>>>(Wn2, wn2pk, 4 * HD, 4, 8 * 4 * 512);

  node_init_kernel<<<(N + 127) / 128, 128, 0, stream>>>(v0, v1, wte, Win0, bin0,
                                                        Win1, bin1, xf, xb, N);
  edge_init_kernel<<<(E + 255) / 256, 256, 0, stream>>>(dist, Wedge, bedge, eb, E);
  fill_u32_kernel<<<8, 256, 0, stream>>>((unsigned*)uf, 0u, NB * HD);
  fill_u32_kernel<<<4, 256, 0, stream>>>((unsigned*)ub, 0u, NB * HD / 2);

  const int etiles = (E + 15) / 16, eblocks = (etiles + 7) / 8;
  const int ntiles = (N + 15) / 16, nblocks = (ntiles + 7) / 8;
  float* out = (float*)d_out;

  for (int layer = 0; layer < NLAYER; ++layer) {
    fill_u32_kernel<<<(N * HD + 255) / 256, 256, 0, stream>>>(agg, 0u, N * HD);
    edge_mlp_kernel<<<eblocks, 256, 0, stream>>>(xb, eb, eidx, eidx + E, welpk,
                                                 bel, agg, E);
    fill_u32_kernel<<<16, 256, 0, stream>>>(xu, 0x007FFFFFu, NB * HD); // enc(-inf)
    node_mlp_kernel<<<nblocks, 256, 0, stream>>>(agg, ub, batch, wn1pk, bn1,
                                                 wn2pk, bn2, xf, xb, xu, N);
    global_update_kernel<<<NB, HD, 0, stream>>>(xu, uf, ub, Wgl, bgl, Wih, bih,
                                                bhh, out + 2 * N + layer * NB * HD);
  }
  out_proj_kernel<<<(N + 255) / 256, 256, 0, stream>>>(xf, Wout0, bout0, Wout1,
                                                       bout1, out, N);
}
